// Compressor_87462714016259
// MI455X (gfx1250) — matrix operand order repaired and checked by reading, not compiled, not run on hardware
//
#include <hip/hip_runtime.h>
#include <hip/hip_bf16.h>
#include <math.h>

// ---- reference constants -------------------------------------------------
#define S_LEN     8192
#define D_MODEL   4096
#define H_DIM     512
#define RATIO     128
#define NC_WIN    64          // S / RATIO
#define RD        64
#define ENTRIES   64
#define NUM_BLOCKS 4
#define EPS       1e-6f
#define HSLICE    128         // channels per workgroup slice
#define KTILE     32          // bf16 WMMA K
#define NTHREADS  512         // 16 waves

typedef __attribute__((ext_vector_type(16))) __bf16 v16bf;
typedef __attribute__((ext_vector_type(8)))  float  v8f;
typedef __attribute__((ext_vector_type(2)))  __bf16 v2bf;

union FragU  { uint4 q[2]; v16bf v; };
union PackU  { v2bf v; unsigned int u; unsigned short s[2]; };

__device__ __forceinline__ unsigned short f2bf(float f) {
    unsigned int u = __float_as_uint(f);
    u += 0x7FFFu + ((u >> 16) & 1u);          // round-to-nearest-even
    return (unsigned short)(u >> 16);
}

// packed f32x2 -> bf16x2 (single v_cvt_pk_bf16_f32 when available)
__device__ __forceinline__ unsigned int f2bf_pk(float a, float b) {
#if __has_builtin(__builtin_amdgcn_cvt_pk_bf16_f32)
    PackU p;
    p.v = __builtin_amdgcn_cvt_pk_bf16_f32(a, b);
    return p.u;
#else
    return (unsigned int)f2bf(a) | ((unsigned int)f2bf(b) << 16);
#endif
}

// A fragment: 16x32 bf16, lane<16 -> K{0..7,16..23}, lane>=16 -> K{8..15,24..31}
__device__ __forceinline__ v16bf frag_a(const unsigned short* aT, int mrow, int lane) {
    const int kb = (lane < 16) ? 0 : 8;
    FragU f;
    f.q[0] = *(const uint4*)(aT + mrow * KTILE + kb);
    f.q[1] = *(const uint4*)(aT + mrow * KTILE + kb + 16);
    return f.v;
}
// B fragment: 32x16 bf16 stored [N][K] in LDS; lane<16 -> K{0..7,16..23}, lane>=16 -> K{8..15,24..31}
__device__ __forceinline__ v16bf frag_b(const unsigned short* bT, int ncol, int lane) {
    const int kb = (lane < 16) ? 0 : 8;
    FragU f;
    f.q[0] = *(const uint4*)(bT + ncol * KTILE + kb);
    f.q[1] = *(const uint4*)(bT + ncol * KTILE + kb + 16);
    return f.v;
}

// ---------------------------------------------------------------------------
// Kernel 1: fused dual-GEMM (kv + gate) + softmax pooling per window slice.
// Grid: (B * NC_WIN * (H_DIM/HSLICE)), Block: 512 threads (16 waves),
// dynamic LDS 128 KB (well under the 320 KB/WGP limit).
// Wave (wm, wn) owns a 32x32 output region: 2x2 tiles x 2 outputs
// = 8 v8f accumulators = 64 VGPRs -> no scratch spills.
// ---------------------------------------------------------------------------
__global__ void __launch_bounds__(NTHREADS, 1)
compress_gemm(const float* __restrict__ x,
              const float* __restrict__ wkv,
              const float* __restrict__ wgate,
              const float* __restrict__ ape,
              float* __restrict__ comp) {
    extern __shared__ char smem[];
    unsigned short* aT   = (unsigned short*)smem;              // 128x32 bf16 (8 KB)
    unsigned short* bkvT = aT   + RATIO * KTILE;               // 128(N) x 32(K) (8 KB)
    unsigned short* bgT  = bkvT + HSLICE * KTILE;              // 8 KB

    const int t    = threadIdx.x;
    const int lane = t & 31;
    const int wave = t >> 5;                 // 0..15
    const int wm   = wave & 3;               // 4 M-groups of 32 rows
    const int wn   = wave >> 2;              // 4 N-groups of 32 cols

    const int wg  = blockIdx.x;
    const int hs  = wg & 3;                  // H slice
    const int win = wg >> 2;                 // b*64 + nc
    const int hsbase = hs * HSLICE;

    v8f acc_kv[2][2], acc_sc[2][2];
    const v8f vzero = {0.f,0.f,0.f,0.f,0.f,0.f,0.f,0.f};
#pragma unroll
    for (int i = 0; i < 2; ++i)
#pragma unroll
        for (int j = 0; j < 2; ++j) { acc_kv[i][j] = vzero; acc_sc[i][j] = vzero; }

    for (int kt = 0; kt < D_MODEL / KTILE; ++kt) {
        // ---- stage x tile (128 x 32) -> bf16 LDS, row-major ----
#pragma unroll
        for (int p = 0; p < 2; ++p) {
            const int u   = t + p * NTHREADS;     // 1024 float4 units
            const int row = u >> 3;
            const int c4  = (u & 7) * 4;
            const size_t g = (size_t)(win * RATIO + row) * D_MODEL + kt * KTILE + c4;
            const float4 v = *(const float4*)(x + g);
            if (kt + 1 < D_MODEL / KTILE)
                __builtin_prefetch(x + g + KTILE, 0, 0);   // global_prefetch next K tile
            *(uint2*)(aT + row * KTILE + c4) =
                make_uint2(f2bf_pk(v.x, v.y), f2bf_pk(v.z, v.w));
        }
        // ---- stage weight tiles (32 x 128) transposed to [N][K] bf16 ----
#pragma unroll
        for (int p = 0; p < 2; ++p) {
            const int u  = t + p * NTHREADS;      // 1024 float4 units
            const int k  = u >> 5;
            const int n4 = (u & 31) * 4;
            const size_t g = (size_t)(kt * KTILE + k) * H_DIM + hsbase + n4;
            const float4 vk = *(const float4*)(wkv + g);
            {
                PackU lo, hi; lo.u = f2bf_pk(vk.x, vk.y); hi.u = f2bf_pk(vk.z, vk.w);
                bkvT[(n4 + 0) * KTILE + k] = lo.s[0];
                bkvT[(n4 + 1) * KTILE + k] = lo.s[1];
                bkvT[(n4 + 2) * KTILE + k] = hi.s[0];
                bkvT[(n4 + 3) * KTILE + k] = hi.s[1];
            }
            const float4 vg = *(const float4*)(wgate + g);
            {
                PackU lo, hi; lo.u = f2bf_pk(vg.x, vg.y); hi.u = f2bf_pk(vg.z, vg.w);
                bgT[(n4 + 0) * KTILE + k] = lo.s[0];
                bgT[(n4 + 1) * KTILE + k] = lo.s[1];
                bgT[(n4 + 2) * KTILE + k] = hi.s[0];
                bgT[(n4 + 3) * KTILE + k] = hi.s[1];
            }
        }
        __syncthreads();

        // ---- WMMA compute: 2 M-tiles x 2 N-tiles x 2 outputs = 8 WMMAs ----
        v16bf af[2];
        af[0] = frag_a(aT, wm * 32 +  0 + (lane & 15), lane);
        af[1] = frag_a(aT, wm * 32 + 16 + (lane & 15), lane);
#pragma unroll
        for (int tn = 0; tn < 2; ++tn) {
            const int ncol = wn * 32 + tn * 16 + (lane & 15);
            const v16bf bk = frag_b(bkvT, ncol, lane);
            const v16bf bg = frag_b(bgT,  ncol, lane);
#pragma unroll
            for (int tm = 0; tm < 2; ++tm) {
                acc_kv[tm][tn] = __builtin_amdgcn_wmma_f32_16x16x32_bf16(
                    false, af[tm], false, bk, (short)0, acc_kv[tm][tn], false, false);
                acc_sc[tm][tn] = __builtin_amdgcn_wmma_f32_16x16x32_bf16(
                    false, af[tm], false, bg, (short)0, acc_sc[tm][tn], false, false);
            }
        }
        __syncthreads();
    }

    // ---- epilogue: spill acc tiles to LDS f32 [128 M][128 N] ----
    float* fkv = (float*)smem;                 // 64 KB
    float* fsc = fkv + RATIO * HSLICE;         // 64 KB
#pragma unroll
    for (int tm = 0; tm < 2; ++tm)
#pragma unroll
        for (int tn = 0; tn < 2; ++tn) {
            const int nn = wn * 32 + tn * 16 + (lane & 15);
#pragma unroll
            for (int r = 0; r < 8; ++r) {
                const int mm = wm * 32 + tm * 16 + r + ((lane < 16) ? 0 : 8);
                fkv[mm * HSLICE + nn] = acc_kv[tm][tn][r];
                fsc[mm * HSLICE + nn] = acc_sc[tm][tn][r];
            }
        }
    __syncthreads();

    // ---- per-channel softmax over the 128-token window, weighted pool ----
    if (t < HSLICE) {
        const int n  = t;
        const int ng = hsbase + n;
        float mx = -INFINITY;
        for (int m = 0; m < RATIO; ++m) {
            const float s = fsc[m * HSLICE + n] + ape[m * H_DIM + ng];
            fsc[m * HSLICE + n] = s;
            mx = fmaxf(mx, s);
        }
        float sum = 0.f, accv = 0.f;
        for (int m = 0; m < RATIO; ++m) {
            const float e = __expf(fsc[m * HSLICE + n] - mx);
            sum  += e;
            accv += e * fkv[m * HSLICE + n];
        }
        comp[(size_t)win * H_DIM + ng] = accv / sum;
    }
}

// ---------------------------------------------------------------------------
// Kernel 2: RMSNorm + RoPE + scatter into paged cache. One block per window.
// ---------------------------------------------------------------------------
__global__ void finalize_scatter(const float* __restrict__ comp,
                                 const float* __restrict__ norm_weight,
                                 const float* __restrict__ cosT,
                                 const float* __restrict__ sinT,
                                 const int*   __restrict__ block_offsets,
                                 float* __restrict__ cache) {
    __shared__ float red[256];
    const int win = blockIdx.x;          // b*64 + nc
    const int b   = win / NC_WIN;
    const int nc  = win % NC_WIN;
    const int t   = threadIdx.x;
    const int h0  = 2 * t, h1 = 2 * t + 1;

    const float c0 = comp[(size_t)win * H_DIM + h0];
    const float c1 = comp[(size_t)win * H_DIM + h1];

    red[t] = c0 * c0 + c1 * c1;
    __syncthreads();
    for (int s = 128; s > 0; s >>= 1) {
        if (t < s) red[t] += red[t + s];
        __syncthreads();
    }
    const float r = rsqrtf(red[0] / (float)H_DIM + EPS);

    float v0 = c0 * r * norm_weight[h0];
    float v1 = c1 * r * norm_weight[h1];

    if (h0 >= H_DIM - RD) {              // RoPE pair at window-start position
        const int i   = t - (H_DIM - RD) / 2;       // 0..31
        const int pos = nc * RATIO;
        const float c = cosT[(size_t)pos * (RD / 2) + i];
        const float s = sinT[(size_t)pos * (RD / 2) + i];
        const float rr = v0 * c - v1 * s;
        const float ri = v0 * s + v1 * c;
        v0 = rr; v1 = ri;
    }

    const int phys = block_offsets[b];   // blk==0 always (nc*128 < BLOCK_SIZE)
    float* dst = cache + (size_t)phys * ENTRIES * H_DIM + (size_t)nc * H_DIM;
    dst[h0] = v0;
    dst[h1] = v1;
}

__global__ void zero_out(float* __restrict__ p, int n) {
    const int i = blockIdx.x * blockDim.x + threadIdx.x;
    if (i < n) p[i] = 0.f;
}

extern "C" void kernel_launch(void* const* d_in, const int* in_sizes, int n_in,
                              void* d_out, int out_size, void* d_ws, size_t ws_size,
                              hipStream_t stream) {
    const float* x     = (const float*)d_in[0];
    const float* wkv   = (const float*)d_in[1];
    const float* wgate = (const float*)d_in[2];
    const float* ape   = (const float*)d_in[3];
    const float* nw    = (const float*)d_in[4];
    const float* cosT  = (const float*)d_in[5];
    const float* sinT  = (const float*)d_in[6];
    const int*   bo    = (const int*)d_in[7];

    const int B = in_sizes[0] / (S_LEN * D_MODEL);
    float* comp = (float*)d_ws;                       // [B*64, 512] f32 scratch
    float* out  = (float*)d_out;                      // [4, 64, 512] f32

    zero_out<<<(out_size + 255) / 256, 256, 0, stream>>>(out, out_size);

    const size_t lds_bytes = 2u * RATIO * HSLICE * sizeof(float);  // 128 KB
    compress_gemm<<<dim3(B * NC_WIN * (H_DIM / HSLICE)), dim3(NTHREADS), lds_bytes, stream>>>(
        x, wkv, wgate, ape, comp);

    finalize_scatter<<<dim3(B * NC_WIN), dim3(256), 0, stream>>>(
        comp, nw, cosT, sinT, bo, out);
}
